// EmbedLoopyBP_77601469104748
// MI455X (gfx1250) — matrix operand order repaired and checked by reading, not compiled, not run on hardware
//
#include <hip/hip_runtime.h>

// Problem constants (match reference setup_inputs()).
#define NN      50000      // nodes
#define NE      800000     // directed edges (2*M), reverse edge = e ^ 1
#define FN_DIM  128
#define FE_DIM  32
#define L_DIM   128
#define O_DIM   128
#define NG      128        // graphs
#define MAX_LV  3

#define WT_STRIDE   136    // padded bf16 row stride for K=128 (bank-conflict-free frags)
#define WT_STRIDE_E 40     // padded bf16 row stride for K=32

typedef __attribute__((ext_vector_type(16))) __bf16 v16bf;
typedef __attribute__((ext_vector_type(8)))  float  v8f;
typedef __attribute__((ext_vector_type(4)))  float  v4f;

__device__ __forceinline__ v8f wmma_bf16(v16bf a, v16bf b, v8f c) {
  // D = A(16x32 bf16) x B(32x16 bf16) + C(16x16 f32)
  return __builtin_amdgcn_wmma_f32_16x16x32_bf16(
      /*neg_a=*/false, a, /*neg_b=*/false, b,
      /*c_mod=*/(short)0, c, /*reuse_a=*/false, /*reuse_b=*/false);
}

__device__ __forceinline__ v4f ldnt4(const float* p) {
  // streaming load: TH=NT, keeps the 192MB L2 free for the gather working set
  return __builtin_nontemporal_load((const v4f*)p);
}
__device__ __forceinline__ v4f ld4(const float* p) {
  return *(const v4f*)p;
}
__device__ __forceinline__ v4f relu4(v4f a) {
  v4f r;
  r.x = fmaxf(a.x, 0.f); r.y = fmaxf(a.y, 0.f);
  r.z = fmaxf(a.z, 0.f); r.w = fmaxf(a.w, 0.f);
  return r;
}

// Pack 16 f32 (two K-groups of 8) into one bf16 A-fragment.
// ISA A layout (16-bit A 16x32): lanes0-15 elems0..7 = K0..7, elems8..15 = K16..23;
// lanes16-31 elems0..7 = K8..15, elems8..15 = K24..31  ->  per-lane K = half*8 (+16).
__device__ __forceinline__ v16bf make_afrag(v4f a0, v4f a1, v4f a2, v4f a3) {
  v16bf a;
  a[0]  = (__bf16)a0.x; a[1]  = (__bf16)a0.y; a[2]  = (__bf16)a0.z; a[3]  = (__bf16)a0.w;
  a[4]  = (__bf16)a1.x; a[5]  = (__bf16)a1.y; a[6]  = (__bf16)a1.z; a[7]  = (__bf16)a1.w;
  a[8]  = (__bf16)a2.x; a[9]  = (__bf16)a2.y; a[10] = (__bf16)a2.z; a[11] = (__bf16)a2.w;
  a[12] = (__bf16)a3.x; a[13] = (__bf16)a3.y; a[14] = (__bf16)a3.z; a[15] = (__bf16)a3.w;
  return a;
}

// B fragment: lane holds column n = lane&15; k0 = base + (lane>>4)*8; elems 0..7 -> K = k0 + i, elems 8..15 -> K = k0 + 16 + (i-8).
// wt is transposed bf16 [Ncol][stride]: two runs of 8 bf16 -> two ds_load_b128.
__device__ __forceinline__ v16bf load_bfrag(const __bf16* wt, int n, int k0, int stride) {
  v16bf b;
  __builtin_memcpy(&b, wt + (size_t)n * stride + k0, 16);
  __builtin_memcpy((char*)&b + 16, wt + (size_t)n * stride + k0 + 16, 16);
  return b;
}

__device__ __forceinline__ void lds_fill(__bf16* lds, const __bf16* g, int n_bf16) {
  const unsigned* gs = (const unsigned*)g;
  unsigned* ls = (unsigned*)lds;
  for (int i = threadIdx.x; i < (n_bf16 >> 1); i += blockDim.x) ls[i] = gs[i];
  __syncthreads();
}

// ---------------- weight prep: W[K][Ncol] f32 -> WT[n][k] bf16 (padded) --------
__global__ void prep_wt_k(const float* __restrict__ W, __bf16* __restrict__ WT,
                          int K, int Ncol, int stride) {
  int i = blockIdx.x * blockDim.x + threadIdx.x;
  if (i >= K * Ncol) return;
  int k = i / Ncol, n = i - k * Ncol;
  WT[(size_t)n * stride + k] = (__bf16)W[(size_t)k * Ncol + n];
}

__global__ void zero_k(float* __restrict__ p, int n) {
  int i = blockIdx.x * blockDim.x + threadIdx.x;
  if (i < n) p[i] = 0.f;
}

// --------- node_lin = node_feat @ W_n2l + b  (N x 128 @ 128 x 128) -------------
__global__ void node_linear_k(const float* __restrict__ A,
                              const __bf16* __restrict__ WTg,
                              const float* __restrict__ bias,
                              float* __restrict__ out, int ntiles) {
  __shared__ __bf16 wt[128 * WT_STRIDE];
  lds_fill(wt, WTg, 128 * WT_STRIDE);
  const int lane = threadIdx.x & 31, wave = threadIdx.x >> 5;
  const int tile = blockIdx.x * 4 + wave;
  if (tile >= ntiles) return;
  const int r = lane & 15, half = lane >> 4;
  const float* arow = A + (size_t)(tile * 16 + r) * 128;

  v16bf afrag[4];
#pragma unroll
  for (int kt = 0; kt < 4; ++kt) {
    int g1 = kt * 32 + half * 8;
    afrag[kt] = make_afrag(ldnt4(arow + g1),      ldnt4(arow + g1 + 4),
                           ldnt4(arow + g1 + 16), ldnt4(arow + g1 + 20));
  }
#pragma unroll
  for (int nt = 0; nt < 8; ++nt) {
    v8f c = {0.f, 0.f, 0.f, 0.f, 0.f, 0.f, 0.f, 0.f};
#pragma unroll
    for (int kt = 0; kt < 4; ++kt)
      c = wmma_bf16(afrag[kt], load_bfrag(wt, nt * 16 + r, kt * 32 + half * 8, WT_STRIDE), c);
    const int col = nt * 16 + r;
    const float bcol = bias[col];
#pragma unroll
    for (int g = 0; g < 8; ++g)   // node_lin is gathered later: keep RT (L2-resident)
      out[(size_t)(tile * 16 + g + half * 8) * 128 + col] = c[g] + bcol;
  }
}

// - input_message = node_lin[src] + edge_feat @ W_e2l + b ; cur = relu(im) ------
__global__ void edge_init_k(const float* __restrict__ EF,
                            const __bf16* __restrict__ WTg,
                            const float* __restrict__ bias,
                            const float* __restrict__ node_lin,
                            const int* __restrict__ edge_src,
                            float* __restrict__ im, float* __restrict__ cur) {
  __shared__ __bf16 wt[128 * WT_STRIDE_E];
  lds_fill(wt, WTg, 128 * WT_STRIDE_E);
  const int lane = threadIdx.x & 31, wave = threadIdx.x >> 5;
  const int tile = blockIdx.x * 4 + wave;
  const int r = lane & 15, half = lane >> 4;
  const float* arow = EF + (size_t)(tile * 16 + r) * FE_DIM;
  const int g1 = half * 8;
  v16bf afrag = make_afrag(ldnt4(arow + g1),      ldnt4(arow + g1 + 4),
                           ldnt4(arow + g1 + 16), ldnt4(arow + g1 + 20));
  int srow[8];
#pragma unroll
  for (int g = 0; g < 8; ++g) srow[g] = edge_src[tile * 16 + g + half * 8];

#pragma unroll
  for (int nt = 0; nt < 8; ++nt) {
    v8f c = {0.f, 0.f, 0.f, 0.f, 0.f, 0.f, 0.f, 0.f};
    c = wmma_bf16(afrag, load_bfrag(wt, nt * 16 + r, half * 8, WT_STRIDE_E), c);
    const int col = nt * 16 + r;
    const float bcol = bias[col];
#pragma unroll
    for (int g = 0; g < 8; ++g) {
      const size_t o = (size_t)(tile * 16 + g + half * 8) * 128 + col;
      float v = c[g] + bcol + node_lin[(size_t)srow[g] * 128 + col];
      __builtin_nontemporal_store(v, &im[o]);               // 410MB stream
      __builtin_nontemporal_store(fmaxf(v, 0.f), &cur[o]);  // 410MB stream
    }
  }
}

// ------- node_agg[dst] += cur  (32 threads of float4 per edge row) -------------
__global__ void scatter_k(const float* __restrict__ cur,
                          const int* __restrict__ edge_dst,
                          float* __restrict__ node_agg) {
  int idx = blockIdx.x * blockDim.x + threadIdx.x;
  int e = idx >> 5, q = idx & 31;
  if (e >= NE) return;
  int d = edge_dst[e];
  v4f v = ldnt4(cur + (size_t)e * 128 + q * 4);  // stream cur, don't evict node_agg
  float* p = node_agg + (size_t)d * 128 + q * 4;
  unsafeAtomicAdd(p + 0, v.x);
  unsafeAtomicAdd(p + 1, v.y);
  unsafeAtomicAdd(p + 2, v.z);
  unsafeAtomicAdd(p + 3, v.w);
}

// -- cur_out = relu((node_agg[src] - cur_in[e^1]) @ conv_W + conv_b + im) -------
__global__ void conv_k(const float* __restrict__ node_agg,
                       const float* __restrict__ cur_in,
                       const __bf16* __restrict__ WTg,
                       const float* __restrict__ bias,
                       const float* __restrict__ im,
                       const int* __restrict__ edge_src,
                       float* __restrict__ cur_out) {
  __shared__ __bf16 wt[128 * WT_STRIDE];
  lds_fill(wt, WTg, 128 * WT_STRIDE);
  const int lane = threadIdx.x & 31, wave = threadIdx.x >> 5;
  const int tile = blockIdx.x * 4 + wave;
  const int r = lane & 15, half = lane >> 4;
  const int e = tile * 16 + r;
  const float* na = node_agg + (size_t)edge_src[e] * 128;  // RT: L2-resident gather
  const float* cr = cur_in + (size_t)(e ^ 1) * 128;        // NT: stream

  v16bf afrag[4];
#pragma unroll
  for (int kt = 0; kt < 4; ++kt) {
    int g1 = kt * 32 + half * 8;
    afrag[kt] = make_afrag(ld4(na + g1)      - ldnt4(cr + g1),
                           ld4(na + g1 + 4)  - ldnt4(cr + g1 + 4),
                           ld4(na + g1 + 16) - ldnt4(cr + g1 + 16),
                           ld4(na + g1 + 20) - ldnt4(cr + g1 + 20));
  }
#pragma unroll
  for (int nt = 0; nt < 8; ++nt) {
    v8f c = {0.f, 0.f, 0.f, 0.f, 0.f, 0.f, 0.f, 0.f};
#pragma unroll
    for (int kt = 0; kt < 4; ++kt)
      c = wmma_bf16(afrag[kt], load_bfrag(wt, nt * 16 + r, kt * 32 + half * 8, WT_STRIDE), c);
    const int col = nt * 16 + r;
    const float bcol = bias[col];
#pragma unroll
    for (int g = 0; g < 8; ++g) {
      const size_t o = (size_t)(tile * 16 + g + half * 8) * 128 + col;
      float v = fmaxf(c[g] + bcol + __builtin_nontemporal_load(&im[o]), 0.f);
      __builtin_nontemporal_store(v, &cur_out[o]);
    }
  }
}

// -- y[gid] += relu(relu(node_agg) @ out_W + out_b), pooled per graph -----------
__global__ void out_pool_k(const float* __restrict__ node_agg,
                           const __bf16* __restrict__ WTg,
                           const float* __restrict__ bias,
                           const int* __restrict__ graph_ids,
                           float* __restrict__ y, int ntiles) {
  __shared__ __bf16 wt[128 * WT_STRIDE];
  lds_fill(wt, WTg, 128 * WT_STRIDE);
  const int lane = threadIdx.x & 31, wave = threadIdx.x >> 5;
  const int tile = blockIdx.x * 4 + wave;
  if (tile >= ntiles) return;
  const int r = lane & 15, half = lane >> 4;
  const float* arow = node_agg + (size_t)(tile * 16 + r) * 128;

  v16bf afrag[4];
#pragma unroll
  for (int kt = 0; kt < 4; ++kt) {
    int g1 = kt * 32 + half * 8;
    afrag[kt] = make_afrag(relu4(ldnt4(arow + g1)),
                           relu4(ldnt4(arow + g1 + 4)),
                           relu4(ldnt4(arow + g1 + 16)),
                           relu4(ldnt4(arow + g1 + 20)));
  }
  int gid[8];
#pragma unroll
  for (int g = 0; g < 8; ++g) gid[g] = graph_ids[tile * 16 + g + half * 8];

#pragma unroll
  for (int nt = 0; nt < 8; ++nt) {
    v8f c = {0.f, 0.f, 0.f, 0.f, 0.f, 0.f, 0.f, 0.f};
#pragma unroll
    for (int kt = 0; kt < 4; ++kt)
      c = wmma_bf16(afrag[kt], load_bfrag(wt, nt * 16 + r, kt * 32 + half * 8, WT_STRIDE), c);
    const int col = nt * 16 + r;
    const float bcol = bias[col];
#pragma unroll
    for (int g = 0; g < 8; ++g)
      unsafeAtomicAdd(&y[(size_t)gid[g] * 128 + col], fmaxf(c[g] + bcol, 0.f));
  }
}

__global__ void relu_copy_k(const float* __restrict__ y, float* __restrict__ out, int n) {
  int i = blockIdx.x * blockDim.x + threadIdx.x;
  if (i < n) out[i] = fmaxf(y[i], 0.f);
}

extern "C" void kernel_launch(void* const* d_in, const int* in_sizes, int n_in,
                              void* d_out, int out_size, void* d_ws, size_t ws_size,
                              hipStream_t stream) {
  const float* node_feat = (const float*)d_in[0];
  const float* edge_feat = (const float*)d_in[1];
  const float* w_n2l_W = (const float*)d_in[2];
  const float* w_n2l_b = (const float*)d_in[3];
  const float* w_e2l_W = (const float*)d_in[4];
  const float* w_e2l_b = (const float*)d_in[5];
  const float* conv_W  = (const float*)d_in[6];
  const float* conv_b  = (const float*)d_in[7];
  const float* out_W   = (const float*)d_in[8];
  const float* out_b   = (const float*)d_in[9];
  const int* edge_src  = (const int*)d_in[10];
  const int* edge_dst  = (const int*)d_in[11];
  const int* graph_ids = (const int*)d_in[12];

  char* ws = (char*)d_ws;
  size_t off = 0;
  auto carve = [&](size_t bytes) -> void* {
    void* p = ws + off;
    off += (bytes + 255) & ~(size_t)255;
    return p;
  };
  float*  node_lin = (float*)carve((size_t)NN * 128 * 4);
  float*  node_agg = (float*)carve((size_t)NN * 128 * 4);
  float*  im       = (float*)carve((size_t)NE * 128 * 4);
  float*  cur0     = (float*)carve((size_t)NE * 128 * 4);
  float*  cur1     = (float*)carve((size_t)NE * 128 * 4);
  float*  ybuf     = (float*)carve((size_t)NG * 128 * 4);
  __bf16* wt_n2l   = (__bf16*)carve((size_t)128 * WT_STRIDE * 2);
  __bf16* wt_conv  = (__bf16*)carve((size_t)128 * WT_STRIDE * 2);
  __bf16* wt_out   = (__bf16*)carve((size_t)128 * WT_STRIDE * 2);
  __bf16* wt_e2l   = (__bf16*)carve((size_t)128 * WT_STRIDE_E * 2);

  // weight prep (f32 -> transposed padded bf16)
  prep_wt_k<<<64, 256, 0, stream>>>(w_n2l_W, wt_n2l, 128, 128, WT_STRIDE);
  prep_wt_k<<<64, 256, 0, stream>>>(conv_W,  wt_conv, 128, 128, WT_STRIDE);
  prep_wt_k<<<64, 256, 0, stream>>>(out_W,   wt_out,  128, 128, WT_STRIDE);
  prep_wt_k<<<16, 256, 0, stream>>>(w_e2l_W, wt_e2l,  32, 128, WT_STRIDE_E);

  const int node_tiles = NN / 16;               // 3125
  const int node_blocks = (node_tiles + 3) / 4; // 782
  const int edge_blocks = NE / 64;              // 12500
  const int zeroN_blocks = (NN * 128 + 255) / 256;
  const int scat_blocks = (NE * 32) / 256;      // 100000

  node_linear_k<<<node_blocks, 128, 0, stream>>>(node_feat, wt_n2l, w_n2l_b,
                                                 node_lin, node_tiles);
  edge_init_k<<<edge_blocks, 128, 0, stream>>>(edge_feat, wt_e2l, w_e2l_b,
                                               node_lin, edge_src, im, cur0);

  float* a = cur0;
  float* b = cur1;
  for (int lv = 0; lv < MAX_LV; ++lv) {
    zero_k<<<zeroN_blocks, 256, 0, stream>>>(node_agg, NN * 128);
    scatter_k<<<scat_blocks, 256, 0, stream>>>(a, edge_dst, node_agg);
    conv_k<<<edge_blocks, 128, 0, stream>>>(node_agg, a, wt_conv, conv_b, im,
                                            edge_src, b);
    float* t = a; a = b; b = t;
  }

  zero_k<<<zeroN_blocks, 256, 0, stream>>>(node_agg, NN * 128);
  scatter_k<<<scat_blocks, 256, 0, stream>>>(a, edge_dst, node_agg);
  zero_k<<<(NG * 128 + 255) / 256, 256, 0, stream>>>(ybuf, NG * 128);
  out_pool_k<<<node_blocks, 128, 0, stream>>>(node_agg, wt_out, out_b, graph_ids,
                                              ybuf, node_tiles);
  relu_copy_k<<<(NG * 128 + 255) / 256, 256, 0, stream>>>(ybuf, (float*)d_out,
                                                          NG * 128);
}
